// Head_28518582845924
// MI455X (gfx1250) — matrix operand order repaired and checked by reading, not compiled, not run on hardware
//
#include <hip/hip_runtime.h>
#include <hip/hip_bf16.h>
#include <math.h>

// ---------------------------------------------------------------------------
// Single-head causal attention, MI455X (gfx1250), wave32, WMMA f16 -> f32,
// x-tile staged via the Tensor Data Mover (tensor_load_to_lds).
//   B=512 batches, T=256 tokens, H=128 head dim.
//   One workgroup (256 threads = 8 waves) per batch element.
// ---------------------------------------------------------------------------

#define TT 256
#define HH 128

typedef __attribute__((ext_vector_type(16))) _Float16 v16h;
typedef __attribute__((ext_vector_type(8)))  float    v8f;
typedef __attribute__((ext_vector_type(4)))  unsigned int u32x4;
typedef __attribute__((ext_vector_type(8)))  int      i32x8;
typedef __attribute__((ext_vector_type(4)))  int      i32x4;

union F16x16 { v16h v; u32x4 u[2]; };

#if defined(__has_builtin)
#  if __has_builtin(__builtin_amdgcn_tensor_load_to_lds)
#    define HAVE_TDM 1
#  else
#    define HAVE_TDM 0
#  endif
#else
#  define HAVE_TDM 0
#endif

__device__ __forceinline__ v8f wmma_f16(v16h a, v16h b, v8f c) {
    // D = A(16x32 f16) * B(32x16 f16) + C(16x16 f32)
    return __builtin_amdgcn_wmma_f32_16x16x32_f16(
        /*neg_a=*/false, a, /*neg_b=*/false, b,
        /*c_mod=*/(short)0, c, /*reuse_a=*/false, /*reuse_b=*/false);
}

// Load a 16x32 f16 A-fragment (or 32x16 B-fragment) from two 16B chunks.
__device__ __forceinline__ v16h load_frag2(const _Float16* p0, const _Float16* p1) {
    F16x16 f;
    f.u[0] = *(const u32x4*)p0;
    f.u[1] = *(const u32x4*)p1;
    return f.v;
}

// ---------------------------------------------------------------------------
// Kernel 0: convert W (f32) -> f16 workspace, order [Q,K,V][e][d].
// ---------------------------------------------------------------------------
__global__ void __launch_bounds__(256)
wconvert_kernel(const float* __restrict__ Wq, const float* __restrict__ Wk,
                const float* __restrict__ Wv, _Float16* __restrict__ wh) {
    int i = blockIdx.x * 256 + threadIdx.x;           // 0 .. 3*128*128-1
    int off = i & (HH * HH - 1);
    const float* src = (i < HH * HH) ? Wq : ((i < 2 * HH * HH) ? Wk : Wv);
    wh[i] = (_Float16)src[off];
}

// ---------------------------------------------------------------------------
// Main kernel.  Dynamic LDS layout (f16 units):
//   xh   [256][128]  @ 0       (64 KB)   x tile, f16
//   qh   [256][128]  @ 32768   (64 KB)   q, row-major   \  overlaid by 128 KB
//   kh   [256][128]  @ 65536   (64 KB)   k, row-major   /  f32 x staging (TDM)
//   vT   [128][256]  @ 98304   (64 KB)   v, transposed
//   stg  [8][512]    @ 131072  ( 8 KB)   per-wave P transpose staging
// ---------------------------------------------------------------------------
__global__ void __launch_bounds__(256, 1)
attn_head_kernel(const float* __restrict__ x,
                 const _Float16* __restrict__ wh,     // [3][128][128] Q,K,V f16
                 const float* __restrict__ bq,
                 const float* __restrict__ bk,
                 const float* __restrict__ bv,
                 float* __restrict__ out) {
    extern __shared__ _Float16 lds[];
    _Float16* xh  = lds;
    _Float16* qh  = lds + 32768;
    _Float16* kh  = lds + 65536;
    _Float16* vT  = lds + 98304;
    _Float16* stg = lds + 131072;
    float*    xf  = (float*)(lds + 32768);  // 128 KB f32 staging (overlays qh+kh)

    const int b    = blockIdx.x;
    const int tid  = threadIdx.x;
    const int wave = tid >> 5;
    const int lane = tid & 31;
    const int hi   = lane >> 4;       // half-wave (0/1)
    const int ln   = lane & 15;       // lane within half

#if HAVE_TDM
    // ------------- Phase A0: TDM DMA x[b] (f32, 128 KB) into LDS ---------
    if (tid == 0) {
        const unsigned long long ga =
            (unsigned long long)(const void*)(x + (size_t)b * (TT * HH));
        // D# group 0: count=1 | lds_addr | 57-bit global_addr | type=2
        u32x4 g0;
        g0[0] = 1u;                                   // count=1, user mode
        g0[1] = 65536u;                               // lds_addr (bytes): xf
        g0[2] = (unsigned)(ga & 0xFFFFFFFFu);         // global_addr[31:0]
        g0[3] = (unsigned)((ga >> 32) & 0x01FFFFFFu)  // global_addr[56:32]
              | (2u << 30);                           // type=2 ("image")
        // D# group 1: 1-D copy, data_size=3 (8B), 16384 elements = 128 KB
        i32x8 g1;
        g1[0] = (int)(3u << 16);                      // data_size=8B
        g1[1] = (int)(16384u << 16);                  // tensor_dim0[15:0]
        g1[2] = (int)(1u << 16);                      // tensor_dim1 = 1
        g1[3] = (int)(16384u << 16);                  // tile_dim0 = 16384
        g1[4] = 0;                                    // tile_dim1 = 0 (1-D)
        g1[5] = 16384;                                // tensor_dim0_stride
        g1[6] = 0;
        g1[7] = 0;
        i32x4 gz = {0, 0, 0, 0};
#if __clang_major__ >= 23
        i32x8 gz8 = {0, 0, 0, 0, 0, 0, 0, 0};
        __builtin_amdgcn_tensor_load_to_lds(g0, g1, gz, gz, gz8, 0);
#else
        __builtin_amdgcn_tensor_load_to_lds(g0, g1, gz, gz, 0);
#endif
        __builtin_amdgcn_s_wait_tensorcnt(0);
    }
    __syncthreads();

    // ------------- Phase A1: convert staged f32 -> f16 into xh -----------
    for (int i = tid; i < (TT * HH) / 4; i += 256) {
        float4 f = *(const float4*)(xf + 4 * i);
        union { _Float16 h[4]; unsigned long long u; } p;
        p.h[0] = (_Float16)f.x; p.h[1] = (_Float16)f.y;
        p.h[2] = (_Float16)f.z; p.h[3] = (_Float16)f.w;
        *(unsigned long long*)(xh + 4 * i) = p.u;
    }
#else
    // ------------- Fallback: vector loads, convert, store to LDS ---------
    {
        const float4* xg = (const float4*)(x + (size_t)b * (TT * HH));
        for (int i = tid; i < (TT * HH) / 4; i += 256) {
            float4 f = xg[i];
            union { _Float16 h[4]; unsigned long long u; } p;
            p.h[0] = (_Float16)f.x; p.h[1] = (_Float16)f.y;
            p.h[2] = (_Float16)f.z; p.h[3] = (_Float16)f.w;
            *(unsigned long long*)(xh + 4 * i) = p.u;
        }
    }
#endif
    __syncthreads();

    // ---------------- Phase B: q/k/v = x @ W^T + b via WMMA -------------
    // 128 (row,col) tiles; each wave computes Q, K, V for its tile so the
    // x A-fragments are loaded once and reused for 12 WMMAs (no divergence).
    for (int t = wave; t < 128; t += 8) {
        const int m0 = (t >> 3) * 16;     // row base (token)
        const int e0 = (t & 7) * 16;      // col base (feature)

        v16h a[4];
        #pragma unroll
        for (int kk = 0; kk < 4; kk++) {
            const int d0 = kk * 32 + hi * 16;
            a[kk] = load_frag2(xh + (m0 + ln) * HH + d0,
                               xh + (m0 + ln) * HH + d0 + 8);
        }

        v8f cq, ck, cv;
        {
            const float q0 = bq[e0 + ln], k0 = bk[e0 + ln], v0 = bv[e0 + ln];
            #pragma unroll
            for (int r = 0; r < 8; r++) { cq[r] = q0; ck[r] = k0; cv[r] = v0; }
        }

        const _Float16* wrq = wh + (size_t)(e0 + ln) * HH;              // Q
        const _Float16* wrk = wrq + (size_t)HH * HH;                    // K
        const _Float16* wrv = wrq + (size_t)2 * HH * HH;                // V
        #pragma unroll
        for (int kk = 0; kk < 4; kk++) {
            const int off = kk * 32 + hi * 16;                // B col = W row
            F16x16 fq, fk, fv;
            fq.u[0] = *(const u32x4*)(wrq + off); fq.u[1] = *(const u32x4*)(wrq + off + 8);
            fk.u[0] = *(const u32x4*)(wrk + off); fk.u[1] = *(const u32x4*)(wrk + off + 8);
            fv.u[0] = *(const u32x4*)(wrv + off); fv.u[1] = *(const u32x4*)(wrv + off + 8);
            cq = wmma_f16(a[kk], fq.v, cq);
            ck = wmma_f16(a[kk], fk.v, ck);
            cv = wmma_f16(a[kk], fv.v, cv);
        }
        // D layout: lane -> col n=ln, VGPR r -> row m0 + r + 8*hi
        #pragma unroll
        for (int r = 0; r < 8; r++) {
            const int m = m0 + r + 8 * hi;
            qh[m * HH + e0 + ln]   = (_Float16)cq[r];
            kh[m * HH + e0 + ln]   = (_Float16)ck[r];
            vT[(e0 + ln) * TT + m] = (_Float16)cv[r];   // transposed
        }
    }
    __syncthreads();

    // ---------------- Phase C: causal flash attention -------------------
    float* og = out + (size_t)b * (TT * HH);
    _Float16* stw = stg + wave * 512;       // 16x32 f16 staging per wave

    for (int qi = wave; qi < 16; qi += 8) { // 2 query tiles per wave
        const int i0 = qi * 16;

        v16h aq[4];
        #pragma unroll
        for (int kk = 0; kk < 4; kk++) {
            const int d0 = kk * 32 + hi * 16;
            aq[kk] = load_frag2(qh + (i0 + ln) * HH + d0,
                                qh + (i0 + ln) * HH + d0 + 8);
        }

        v8f o[8];                            // 16x128 f32 output accumulators
        #pragma unroll
        for (int nt = 0; nt < 8; nt++)
            #pragma unroll
            for (int r = 0; r < 8; r++) o[nt][r] = 0.0f;
        float mrow[8], lrow[8];
        #pragma unroll
        for (int r = 0; r < 8; r++) { mrow[r] = -__builtin_inff(); lrow[r] = 0.0f; }

        // Key tiles in pairs of 16 -> K=32 for the PV WMMA.
        for (int kp = 0; kp <= (qi >> 1); kp++) {
            const int j0 = kp * 32;

            v8f s0, s1;
            #pragma unroll
            for (int r = 0; r < 8; r++) { s0[r] = 0.0f; s1[r] = 0.0f; }
            #pragma unroll
            for (int kk = 0; kk < 4; kk++) {
                const int db = kk * 32 + hi * 16;
                const _Float16* k0 = kh + (j0 + ln) * HH + db;    // B col j = k row j
                const _Float16* k1 = kh + (j0 + 16 + ln) * HH + db;
                F16x16 fb0, fb1;
                fb0.u[0] = *(const u32x4*)k0; fb0.u[1] = *(const u32x4*)(k0 + 8);
                fb1.u[0] = *(const u32x4*)k1; fb1.u[1] = *(const u32x4*)(k1 + 8);
                s0 = wmma_f16(aq[kk], fb0.v, s0);
                s1 = wmma_f16(aq[kk], fb1.v, s1);
            }

            // Causal mask (row m = i0 + r + 8*hi, col j = j0 + {0,16} + ln)
            #pragma unroll
            for (int r = 0; r < 8; r++) {
                const int gi = i0 + r + 8 * hi;
                s0[r] = (j0 + ln      > gi) ? -__builtin_inff() : s0[r];
                s1[r] = (j0 + 16 + ln > gi) ? -__builtin_inff() : s1[r];
            }

            // Online softmax: a C-layout row spans 16 lanes of a half-wave.
            #pragma unroll
            for (int r = 0; r < 8; r++) {
                float t0 = fmaxf(s0[r], s1[r]);
                t0 = fmaxf(t0, __shfl_xor(t0, 1, 16));
                t0 = fmaxf(t0, __shfl_xor(t0, 2, 16));
                t0 = fmaxf(t0, __shfl_xor(t0, 4, 16));
                t0 = fmaxf(t0, __shfl_xor(t0, 8, 16));
                const float mn    = fmaxf(mrow[r], t0);
                const float alpha = __expf(mrow[r] - mn);
                const float p0    = __expf(s0[r] - mn);
                const float p1    = __expf(s1[r] - mn);
                float rs = p0 + p1;
                rs += __shfl_xor(rs, 1, 16);
                rs += __shfl_xor(rs, 2, 16);
                rs += __shfl_xor(rs, 4, 16);
                rs += __shfl_xor(rs, 8, 16);
                lrow[r] = lrow[r] * alpha + rs;
                mrow[r] = mn;
                #pragma unroll
                for (int nt = 0; nt < 8; nt++) o[nt][r] *= alpha;
                const int m = r + 8 * hi;
                stw[m * 32 + ln]      = (_Float16)p0;
                stw[m * 32 + 16 + ln] = (_Float16)p1;
            }

            // Reload P as a 16x32 A-fragment (DS ops are in-order per wave).
            v16h ap = load_frag2(stw + ln * 32 + hi * 16,
                                 stw + ln * 32 + hi * 16 + 8);

            // O += P @ V ; B col d = vT row d (contiguous).
            #pragma unroll
            for (int nt = 0; nt < 8; nt++) {
                const _Float16* vp = vT + (nt * 16 + ln) * TT + j0 + hi * 16;
                F16x16 fb;
                fb.u[0] = *(const u32x4*)vp;
                fb.u[1] = *(const u32x4*)(vp + 8);
                o[nt] = wmma_f16(ap, fb.v, o[nt]);
            }
        }

        // Normalize and store f32 output.
        #pragma unroll
        for (int r = 0; r < 8; r++) {
            const float inv = 1.0f / lrow[r];
            const int m = i0 + r + 8 * hi;
            #pragma unroll
            for (int nt = 0; nt < 8; nt++)
                og[m * HH + nt * 16 + ln] = o[nt][r] * inv;
        }
    }
}

// ---------------------------------------------------------------------------
extern "C" void kernel_launch(void* const* d_in, const int* in_sizes, int n_in,
                              void* d_out, int out_size, void* d_ws, size_t ws_size,
                              hipStream_t stream) {
    (void)in_sizes; (void)n_in; (void)out_size; (void)ws_size;
    const float* x  = (const float*)d_in[0];
    const float* Wk = (const float*)d_in[1];
    const float* bk = (const float*)d_in[2];
    const float* Wq = (const float*)d_in[3];
    const float* bq = (const float*)d_in[4];
    const float* Wv = (const float*)d_in[5];
    const float* bv = (const float*)d_in[6];
    float* out = (float*)d_out;

    _Float16* wh = (_Float16*)d_ws;   // 3*128*128 f16 = 96 KB of workspace

    wconvert_kernel<<<(3 * HH * HH) / 256, 256, 0, stream>>>(Wq, Wk, Wv, wh);

    const size_t lds_bytes = (size_t)(131072 + 8 * 512) * sizeof(_Float16); // ~264 KB
    attn_head_kernel<<<512, 256, lds_bytes, stream>>>(x, wh, bq, bk, bv, out);
}
